// PixelFocusAttention_18339510353991
// MI455X (gfx1250) — matrix operand order repaired and checked by reading, not compiled, not run on hardware
//
#include <hip/hip_runtime.h>
#include <math.h>
#include <stdint.h>

// ---------------- problem constants ----------------
constexpr int B_  = 8;
constexpr int N_  = 2000;
constexpr int C_  = 512;
constexpr int H_  = 8;
constexpr int HD_ = 64;
constexpr int W_  = 5;
constexpr int M_  = 1000;           // N / SR
constexpr int KSTR = 3 * C_;        // Y1 row stride in halfs (q | k | v)
constexpr float SCALE_ = 0.125f;    // hd^-0.5
constexpr float LN_EPS_ = 1e-5f;

typedef _Float16 f16x16 __attribute__((ext_vector_type(16)));
typedef float    f32x8  __attribute__((ext_vector_type(8)));
typedef _Float16 h4     __attribute__((ext_vector_type(4)));   // 8B
typedef _Float16 h8     __attribute__((ext_vector_type(8)));   // 16B
typedef float    f4     __attribute__((ext_vector_type(4)));   // 16B

union AFrag { f16x16 v; _Float16 h[16]; };
union BFrag { f16x16 v; _Float16 h[16]; };
union CFrag { f32x8  v; float    f[8];  };

static __device__ __forceinline__ f32x8 wmma_f16(const f16x16& a, const f16x16& b, const f32x8& c) {
    return __builtin_amdgcn_wmma_f32_16x16x32_f16(false, a, false, b, (short)0, c, false, false);
}

// Async global->LDS 16-byte copy (gfx1250, tracked by ASYNCcnt).
// LDS operand is the wave-relative LDS byte offset = low 32 bits of the generic pointer.
static __device__ __forceinline__ void async_b128(void* lds_dst, const void* gsrc) {
    uint32_t lo = (uint32_t)(uintptr_t)lds_dst;
    uint64_t ga = (uint64_t)(uintptr_t)gsrc;
    asm volatile("global_load_async_to_lds_b128 %0, %1, off"
                 :: "v"(lo), "v"(ga) : "memory");
}
static __device__ __forceinline__ void wait_async_le4() {
    asm volatile("s_wait_asynccnt 4" ::: "memory");
}
static __device__ __forceinline__ void wait_async_0() {
    asm volatile("s_wait_asynccnt 0" ::: "memory");
}

// ======================================================================
// Generic GEMM:  Out[m, n] = (TO) ( sum_k A[m,k] * Wt[n,k] + bias[n] )
// A: (Mtot x K) TA (fp32 or f16), lda row stride. Wt: (Nout x K) fp32.
// Tile 64(M) x 32(N) x 32(K), 128 threads = 4 waves.
// REQUIRES: Mtot % 64 == 0, Nout % 32 == 0, K % 32 == 0 (true for all uses).
// ======================================================================
template <typename TA, typename TO>
__global__ __launch_bounds__(128) void gemm_wmma_kernel(
    const TA* __restrict__ A, int lda,
    const float* __restrict__ Wt,
    const float* __restrict__ bias,
    TO* __restrict__ Out, int ldo, int K)
{
    __shared__ _Float16 As[64][36];
    __shared__ _Float16 Bs[32][36];

    const int bm   = blockIdx.x * 64;
    const int bn   = blockIdx.y * 32;
    const int tid  = threadIdx.x;
    const int wave = tid >> 5;
    const int lane = tid & 31;
    const int lcol = lane & 15;
    const int hi   = (lane >= 16);

    CFrag acc0, acc1;
    #pragma unroll
    for (int i = 0; i < 8; ++i) { acc0.f[i] = 0.f; acc1.f[i] = 0.f; }

    const int mloc = wave * 16 + lcol;

    for (int k0 = 0; k0 < K; k0 += 32) {
        // ---- stage A tile (64 x 32) -> f16, vectorized, branch-free ----
        if constexpr (sizeof(TA) == 4) {
            #pragma unroll
            for (int it = 0; it < 4; ++it) {
                int idx = tid + it * 128;            // 512 chunks of 4 floats
                int r = idx >> 3, c4 = (idx & 7) * 4;
                f4 v = *(const f4*)((const float*)A + (size_t)(bm + r) * lda + k0 + c4);
                h4 hv = { (_Float16)v.x, (_Float16)v.y, (_Float16)v.z, (_Float16)v.w };
                *(h4*)&As[r][c4] = hv;
            }
        } else {
            #pragma unroll
            for (int it = 0; it < 2; ++it) {
                int idx = tid + it * 128;            // 256 chunks of 8 halfs
                int r = idx >> 2, c8 = (idx & 3) * 8;
                h8 v = *(const h8*)((const _Float16*)A + (size_t)(bm + r) * lda + k0 + c8);
                h4 lo = { v[0], v[1], v[2], v[3] };
                h4 hi4 = { v[4], v[5], v[6], v[7] };
                *(h4*)&As[r][c8]     = lo;
                *(h4*)&As[r][c8 + 4] = hi4;
            }
        }
        // ---- stage Wt tile (32 x 32) -> f16 ----
        #pragma unroll
        for (int it = 0; it < 2; ++it) {
            int idx = tid + it * 128;                // 256 chunks of 4 floats
            int r = idx >> 3, c4 = (idx & 7) * 4;
            f4 v = *(const f4*)(Wt + (size_t)(bn + r) * K + k0 + c4);
            h4 hv = { (_Float16)v.x, (_Float16)v.y, (_Float16)v.z, (_Float16)v.w };
            *(h4*)&Bs[r][c4] = hv;
        }
        __syncthreads();

        // A fragment (16x32, contiguous k = 16*hi + s, same order as B)
        AFrag a;
        #pragma unroll
        for (int vv = 0; vv < 8; ++vv) {
            int kb = (hi ? 16 : 0) + vv * 2;
            a.h[2 * vv]     = As[mloc][kb];
            a.h[2 * vv + 1] = As[mloc][kb + 1];
        }
        // B fragments (32x16): K = h + 16*hi, N = lane%16
        BFrag b0, b1;
        #pragma unroll
        for (int hh = 0; hh < 16; ++hh) {
            int kd = hh + (hi ? 16 : 0);
            b0.h[hh] = Bs[lcol][kd];
            b1.h[hh] = Bs[16 + lcol][kd];
        }
        acc0.v = wmma_f16(a.v, b0.v, acc0.v);
        acc1.v = wmma_f16(a.v, b1.v, acc1.v);
        __syncthreads();
    }

    #pragma unroll
    for (int r = 0; r < 8; ++r) {
        int mrow = bm + wave * 16 + r + (hi ? 8 : 0);
        int n0 = bn + lcol;
        int n1 = n0 + 16;
        Out[(size_t)mrow * ldo + n0] = (TO)(acc0.f[r] + bias[n0]);
        Out[(size_t)mrow * ldo + n1] = (TO)(acc1.f[r] + bias[n1]);
    }
}

// ======================================================================
// Average pool over SR=2:  xp[b,m,c] = 0.5*(x[b,2m,c] + x[b,2m+1,c])
// ======================================================================
__global__ void pool_kernel(const float* __restrict__ x, float* __restrict__ xp)
{
    int i = blockIdx.x * blockDim.x + threadIdx.x;
    if (i >= B_ * M_ * C_) return;
    int c  = i % C_;
    int bm = i / C_;
    int m  = bm % M_;
    int b  = bm / M_;
    const float* row = x + ((size_t)b * N_ + 2 * m) * C_ + c;
    xp[i] = 0.5f * (row[0] + row[C_]);
}

// ======================================================================
// LayerNorm (population var) + exact GELU, in place. One block per row.
// ======================================================================
__global__ __launch_bounds__(256) void ln_gelu_kernel(
    float* __restrict__ xp, const float* __restrict__ g, const float* __restrict__ bta)
{
    float* p = xp + (size_t)blockIdx.x * C_;
    int t = threadIdx.x;
    float v0 = p[t], v1 = p[t + 256];
    float s = v0 + v1, q = v0 * v0 + v1 * v1;
    #pragma unroll
    for (int off = 16; off >= 1; off >>= 1) {
        s += __shfl_xor(s, off, 32);
        q += __shfl_xor(q, off, 32);
    }
    __shared__ float ss[8], qq[8];
    if ((t & 31) == 0) { ss[t >> 5] = s; qq[t >> 5] = q; }
    __syncthreads();
    if (t == 0) {
        float S = 0.f, Q = 0.f;
        for (int i = 0; i < 8; ++i) { S += ss[i]; Q += qq[i]; }
        ss[0] = S; qq[0] = Q;
    }
    __syncthreads();
    float mu   = ss[0] * (1.f / C_);
    float var  = qq[0] * (1.f / C_) - mu * mu;
    float rstd = rsqrtf(var + LN_EPS_);
    float y0 = (v0 - mu) * rstd * g[t] + bta[t];
    float y1 = (v1 - mu) * rstd * g[t + 256] + bta[t + 256];
    p[t]       = 0.5f * y0 * (1.f + erff(y0 * 0.70710678118f));
    p[t + 256] = 0.5f * y1 * (1.f + erff(y1 * 0.70710678118f));
}

// ======================================================================
// Fused flash attention over [5 local keys | 1000 pooled keys], f16 I/O.
// Block = 128 threads (4 waves); wave w owns 16 queries; grid (32, H, B).
// K/V pooled tiles are double-buffered and streamed with async b128
// global->LDS copies (ASYNCcnt), overlapped with WMMA compute.
// NOTE: tail tiles issue async reads past the logical tensor; those land in
// later (allocated) workspace regions and their scores are masked to -inf.
// ======================================================================
__global__ __launch_bounds__(128) void attn_kernel(
    const _Float16* __restrict__ Y1, const _Float16* __restrict__ kvp,
    _Float16* __restrict__ out)
{
    __shared__ _Float16 qs[64][72];
    __shared__ _Float16 ks[2][32][72];
    __shared__ _Float16 vs[2][32][72];
    __shared__ _Float16 ps[4][16][36];
    __shared__ float    lS[4][16][W_];

    const int tile = blockIdx.x;
    const int h    = blockIdx.y;
    const int b    = blockIdx.z;
    const int tid  = threadIdx.x;
    const int wave = tid >> 5;
    const int lane = tid & 31;
    const int lcol = lane & 15;
    const int hi   = (lane >= 16);
    const int q0   = tile * 64;

    const _Float16* qbase  = Y1 + (size_t)b * N_ * KSTR + h * HD_;           // + n*KSTR + d
    const _Float16* kvbase = kvp + (size_t)b * M_ * (2 * C_) + h * HD_;      // + key*1024 + d (v at +512)

    // ---- issue async stage of q tile (64 x 64 halfs, 512 x b128) ----
    #pragma unroll
    for (int it = 0; it < 4; ++it) {
        int idx = tid + it * 128;
        int r = idx >> 3, c8 = (idx & 7) * 8;
        async_b128(&qs[r][c8], qbase + (size_t)(q0 + r) * KSTR + c8);
    }
    // ---- issue async stage of K/V tile 0 (4 x b128 per thread) ----
    {
        #pragma unroll
        for (int it = 0; it < 2; ++it) {
            int idx = tid + it * 128;
            int kk = idx >> 3, c8 = (idx & 7) * 8;
            const _Float16* gk = kvbase + (size_t)kk * (2 * C_) + c8;
            async_b128(&ks[0][kk][c8], gk);
            async_b128(&vs[0][kk][c8], gk + C_);
        }
    }
    wait_async_le4();          // q complete (K/V tile 0 may still be in flight)
    __syncthreads();

    // ---- local window scores (16 q x 5 k per wave), overlapped with tile-0 DMA ----
    for (int i = lane; i < 16 * W_; i += 32) {
        int qi = i / W_, w = i % W_;
        int n  = q0 + wave * 16 + qi;
        int nc = (n < N_) ? n : (N_ - 1);
        int g  = nc / W_;
        const _Float16* kl = Y1 + ((size_t)b * N_ + g * W_ + w) * KSTR + C_ + h * HD_;
        float sacc = 0.f;
        #pragma unroll
        for (int j = 0; j < HD_ / 4; ++j) {
            h4 qv = *(const h4*)&qs[wave * 16 + qi][4 * j];
            h4 kv = *(const h4*)(kl + 4 * j);
            sacc += (float)qv[0] * (float)kv[0] + (float)qv[1] * (float)kv[1]
                  + (float)qv[2] * (float)kv[2] + (float)qv[3] * (float)kv[3];
        }
        lS[wave][qi][w] = sacc * SCALE_;
    }
    __syncthreads();

    // ---- init online softmax from local part ----
    float rmax[8], rsum[8];
    CFrag Of[4];
    #pragma unroll
    for (int c = 0; c < 4; ++c)
        #pragma unroll
        for (int r = 0; r < 8; ++r) Of[c].f[r] = 0.f;

    #pragma unroll
    for (int r = 0; r < 8; ++r) {
        int row = r + (hi ? 8 : 0);
        float sc[W_], pw[W_];
        float m0 = -INFINITY;
        #pragma unroll
        for (int w = 0; w < W_; ++w) { sc[w] = lS[wave][row][w]; m0 = fmaxf(m0, sc[w]); }
        float sum = 0.f;
        #pragma unroll
        for (int w = 0; w < W_; ++w) { pw[w] = expf(sc[w] - m0); sum += pw[w]; }
        rmax[r] = m0; rsum[r] = sum;
        int n  = q0 + wave * 16 + row;
        int nc = (n < N_) ? n : (N_ - 1);
        int g  = nc / W_;
        const _Float16* vl = Y1 + ((size_t)b * N_ + g * W_) * KSTR + 2 * C_ + h * HD_;
        #pragma unroll
        for (int c = 0; c < 4; ++c) {
            int d = c * 16 + lcol;
            float acc = 0.f;
            #pragma unroll
            for (int w = 0; w < W_; ++w) acc += pw[w] * (float)vl[(size_t)w * KSTR + d];
            Of[c].f[r] = acc;
        }
    }

    // ---- q A-fragments (loop invariant) ----
    const int mloc = wave * 16 + lcol;
    AFrag aq0, aq1;
    #pragma unroll
    for (int vv = 0; vv < 8; ++vv) {
        int kb = (hi ? 16 : 0) + vv * 2;
        aq0.h[2 * vv]     = qs[mloc][kb];
        aq0.h[2 * vv + 1] = qs[mloc][kb + 1];
        aq1.h[2 * vv]     = qs[mloc][32 + kb];
        aq1.h[2 * vv + 1] = qs[mloc][32 + kb + 1];
    }

    // ---- stream pooled keys: double-buffered async tiles of 32 ----
    constexpr int T = (M_ + 31) / 32;                 // 32 tiles
    for (int kt = 0; kt < T; ++kt) {
        int buf = kt & 1;
        if (kt + 1 < T) {
            int nb = buf ^ 1;
            int km1 = (kt + 1) * 32;
            #pragma unroll
            for (int it = 0; it < 2; ++it) {
                int idx = tid + it * 128;
                int kk = idx >> 3, c8 = (idx & 7) * 8;
                const _Float16* gk = kvbase + (size_t)(km1 + kk) * (2 * C_) + c8;
                async_b128(&ks[nb][kk][c8], gk);
                async_b128(&vs[nb][kk][c8], gk + C_);
            }
            wait_async_le4();                         // current tile complete
        } else {
            wait_async_0();
        }
        __syncthreads();

        int km = kt * 32;
        // S(16x32) = q(16x64) @ k^T(64x32)
        BFrag bk00, bk01, bk10, bk11;
        #pragma unroll
        for (int hh = 0; hh < 16; ++hh) {
            int kd = hh + (hi ? 16 : 0);
            bk00.h[hh] = ks[buf][lcol][kd];
            bk01.h[hh] = ks[buf][lcol][32 + kd];
            bk10.h[hh] = ks[buf][16 + lcol][kd];
            bk11.h[hh] = ks[buf][16 + lcol][32 + kd];
        }
        CFrag S0, S1;
        #pragma unroll
        for (int i = 0; i < 8; ++i) { S0.f[i] = 0.f; S1.f[i] = 0.f; }
        S0.v = wmma_f16(aq0.v, bk00.v, S0.v);
        S0.v = wmma_f16(aq1.v, bk01.v, S0.v);
        S1.v = wmma_f16(aq0.v, bk10.v, S1.v);
        S1.v = wmma_f16(aq1.v, bk11.v, S1.v);

        // online softmax per query row
        #pragma unroll
        for (int r = 0; r < 8; ++r) {
            float s0 = (km + lcol      < M_) ? S0.f[r] * SCALE_ : -INFINITY;
            float s1 = (km + 16 + lcol < M_) ? S1.f[r] * SCALE_ : -INFINITY;
            float mx = fmaxf(s0, s1);
            #pragma unroll
            for (int off = 1; off < 16; off <<= 1) mx = fmaxf(mx, __shfl_xor(mx, off, 16));
            float newm = fmaxf(rmax[r], mx);
            float corr = expf(rmax[r] - newm);
            float p0 = expf(s0 - newm);
            float p1 = expf(s1 - newm);
            float psum = p0 + p1;
            #pragma unroll
            for (int off = 1; off < 16; off <<= 1) psum += __shfl_xor(psum, off, 16);
            rsum[r] = rsum[r] * corr + psum;
            rmax[r] = newm;
            #pragma unroll
            for (int c = 0; c < 4; ++c) Of[c].f[r] *= corr;
            int row = r + (hi ? 8 : 0);
            ps[wave][row][lcol]      = (_Float16)p0;
            ps[wave][row][16 + lcol] = (_Float16)p1;
        }
        __syncthreads();

        // O += P(16x32) @ V(32x64)
        AFrag ap;
        #pragma unroll
        for (int vv = 0; vv < 8; ++vv) {
            int kb = (hi ? 16 : 0) + vv * 2;
            ap.h[2 * vv]     = ps[wave][lcol][kb];
            ap.h[2 * vv + 1] = ps[wave][lcol][kb + 1];
        }
        #pragma unroll
        for (int c = 0; c < 4; ++c) {
            BFrag bv;
            #pragma unroll
            for (int hh = 0; hh < 16; ++hh) {
                int kd = hh + (hi ? 16 : 0);
                bv.h[hh] = vs[buf][kd][c * 16 + lcol];
            }
            Of[c].v = wmma_f16(ap.v, bv.v, Of[c].v);
        }
        __syncthreads();
    }

    // ---- finalize & store f16 (B, N, C) head slice ----
    #pragma unroll
    for (int r = 0; r < 8; ++r) {
        int row = r + (hi ? 8 : 0);
        int n = q0 + wave * 16 + row;
        if (n < N_) {
            float inv = 1.f / rsum[r];
            _Float16* o = out + ((size_t)b * N_ + n) * C_ + h * HD_;
            #pragma unroll
            for (int c = 0; c < 4; ++c) o[c * 16 + lcol] = (_Float16)(Of[c].f[r] * inv);
        }
    }
}

// ======================================================================
// Host launch
// ======================================================================
extern "C" void kernel_launch(void* const* d_in, const int* in_sizes, int n_in,
                              void* d_out, int out_size, void* d_ws, size_t ws_size,
                              hipStream_t stream)
{
    const float* x    = (const float*)d_in[0];
    const float* Wq   = (const float*)d_in[1];
    const float* bq   = (const float*)d_in[2];
    const float* Wkv  = (const float*)d_in[3];
    const float* bkv  = (const float*)d_in[4];
    const float* Wc   = (const float*)d_in[5];
    const float* bc   = (const float*)d_in[6];
    const float* ln_g = (const float*)d_in[7];
    const float* ln_b = (const float*)d_in[8];
    const float* Wp   = (const float*)d_in[9];
    const float* bp   = (const float*)d_in[10];

    constexpr size_t BN = (size_t)B_ * N_;   // 16000
    constexpr size_t BM = (size_t)B_ * M_;   // 8000

    // workspace layout (bytes); order matters: async tail reads from Y1/kvp
    // spill into the region that follows them (always inside d_ws).
    constexpr size_t off_Y1    = 0;                                   // f16 BN x 1536
    constexpr size_t off_xpool = off_Y1 + BN * KSTR * 2;              // f32 BM x 512
    constexpr size_t off_xpg   = off_xpool + BM * C_ * 4;             // f32 BM x 512
    constexpr size_t off_kvp   = off_xpg + BM * C_ * 4;               // f16 BM x 1024
    constexpr size_t off_attn  = off_kvp + BM * (2 * C_) * 2;         // f16 BN x 512

    char* wsb = (char*)d_ws;
    _Float16* Y1   = (_Float16*)(wsb + off_Y1);
    float*    xpool= (float*)   (wsb + off_xpool);
    float*    xpg  = (float*)   (wsb + off_xpg);
    _Float16* kvp  = (_Float16*)(wsb + off_kvp);
    _Float16* attn = (_Float16*)(wsb + off_attn);

    dim3 blk(128);

    // q and kv projections into Y1 (f16)
    gemm_wmma_kernel<float, _Float16><<<dim3((int)(BN / 64), C_ / 32), blk, 0, stream>>>(
        x, C_, Wq, bq, Y1, KSTR, C_);
    gemm_wmma_kernel<float, _Float16><<<dim3((int)(BN / 64), (2 * C_) / 32), blk, 0, stream>>>(
        x, C_, Wkv, bkv, Y1 + C_, KSTR, C_);

    // pooled path
    pool_kernel<<<(int)((BM * C_ + 255) / 256), 256, 0, stream>>>(x, xpool);
    gemm_wmma_kernel<float, float><<<dim3((int)(BM / 64), C_ / 32), blk, 0, stream>>>(
        xpool, C_, Wc, bc, xpg, C_, C_);
    ln_gelu_kernel<<<(int)BM, 256, 0, stream>>>(xpg, ln_g, ln_b);
    gemm_wmma_kernel<float, _Float16><<<dim3((int)(BM / 64), (2 * C_) / 32), blk, 0, stream>>>(
        xpg, C_, Wkv, bkv, kvp, 2 * C_, C_);

    // fused flash attention (f16 in, f16 out)
    attn_kernel<<<dim3((N_ + 63) / 64, H_, B_), blk, 0, stream>>>(Y1, kvp, attn);

    // output projection (f16 A, f32 out)
    gemm_wmma_kernel<_Float16, float><<<dim3((int)(BN / 64), C_ / 32), blk, 0, stream>>>(
        attn, C_, Wp, bp, (float*)d_out, C_, C_);
}
